// LinearAttention_85435489452564
// MI455X (gfx1250) — compile-verified, matrix operand order checked by reading, not run on hardware
//
#include <hip/hip_runtime.h>
#include <hip/hip_bf16.h>

typedef __attribute__((ext_vector_type(16))) _Float16 v16h;
typedef __attribute__((ext_vector_type(8)))  _Float16 v8h;
typedef __attribute__((ext_vector_type(8)))  float    v8f;

#define B_ 2
#define F_ 512
#define S_ 1024
#define E_ 4
#define I_ 1024
#define TOK (B_*S_)
#define EPSN 1e-5f
#define SLOPE_ 0.02f

// Fragment layout helper: 16-bit A/B operand mapping (ISA 7.12.2).
// lane(l), j in [0,16): k-offset within a 32-K tile = khalf + (j<8 ? j : j+8),
// khalf = (l>=16)*8; row/col = l&15.
__device__ __forceinline__ int frag_j(int kin) {
  return kin < 8 ? kin : (kin < 16 ? kin - 8 : (kin < 24 ? kin - 8 : kin - 16));
}
__device__ __forceinline__ int frag_lane(int kin, int n) {
  return n + (((kin >> 3) & 1) ? 16 : 0);
}

// ---------------- weight pre-pack: f32 -> f16 in WMMA fragment order --------
// Output element idx = (((e*MT + mt)*KT + kt)*32 + lane)*16 + j
// mode 0: W[((e*Cin)+k)*Cout + m]   (w0, w2: [E,Cin,Cout])
// mode 1: W[m*Cin + k]              (w1: [Cout rows, Cin], E must be 1)
__global__ void pack_weights_kernel(const float* __restrict__ W,
                                    _Float16* __restrict__ Wp,
                                    int E, int Cin, int Cout, int mode) {
  size_t idx = (size_t)blockIdx.x * blockDim.x + threadIdx.x;
  size_t total = (size_t)E * Cin * Cout;
  if (idx >= total) return;
  int j    = (int)(idx & 15);
  int lane = (int)((idx >> 4) & 31);
  size_t rest = idx >> 9;
  int KT = Cin >> 5;
  int kt = (int)(rest % KT); rest /= KT;
  int MT = Cout >> 4;
  int mt = (int)(rest % MT);
  int e  = (int)(rest / MT);
  int khalf = (lane >> 4) * 8;
  int k = kt * 32 + khalf + (j < 8 ? j : j + 8);
  int m = mt * 16 + (lane & 15);
  float v = mode ? W[(size_t)m * Cin + k]
                 : W[((size_t)e * Cin + k) * Cout + m];
  Wp[idx] = (_Float16)v;
}

__device__ __forceinline__ v16h load_frag(const _Float16* p) {
  v8h lo = *(const v8h*)p;
  v8h hi = *(const v8h*)(p + 8);
  v16h r;
#pragma unroll
  for (int j = 0; j < 8; ++j) { r[j] = lo[j]; r[j + 8] = hi[j]; }
  return r;
}

// ---------------- gating: logits -> softmax -> argmax, loss accumulation ----
__global__ void gate_kernel(const float* __restrict__ x, int C,
                            const float* __restrict__ gw,
                            int* __restrict__ eidx,
                            float* __restrict__ lacc) {
  extern __shared__ float gws[];
  for (int p = threadIdx.x; p < E_ * C; p += blockDim.x) gws[p] = gw[p];
  __syncthreads();
  int t = blockIdx.x * blockDim.x + threadIdx.x;
  if (t >= TOK) return;
  int b = t / S_, s = t % S_;
  const float* xb = x + (size_t)b * C * S_ + s;
  float a0 = 0.f, a1 = 0.f, a2 = 0.f, a3 = 0.f;
  for (int c = 0; c < C; ++c) {
    float xv = xb[(size_t)c * S_];
    a0 += xv * gws[0 * C + c];
    a1 += xv * gws[1 * C + c];
    a2 += xv * gws[2 * C + c];
    a3 += xv * gws[3 * C + c];
  }
  float a[4] = {a0, a1, a2, a3};
  int best = 0; float m = a[0];
  for (int e = 1; e < E_; ++e) { if (a[e] > m) { m = a[e]; best = e; } }
  float g[4]; float se = 0.f;
  for (int e = 0; e < E_; ++e) { g[e] = __expf(a[e] - m); se += g[e]; }
  float inv = 1.0f / se;
  eidx[t] = best;
  for (int e = 0; e < E_; ++e) atomicAdd(&lacc[e], g[e] * inv);
  atomicAdd(&lacc[4 + best], 1.0f);
}

// ---------------- routed GEMM via dense WMMA, packed-f16 weights ------------
// Block: 256 thr = 8 waves; each wave owns TWO 16x16 tiles (B-fragment reuse).
// grid.x = Cout/256, grid.y = TOK/16. X:[B,Cin,S] f32, Wp packed fragments.
__global__ void moe_wmma_kernel(const float* __restrict__ X, int Cin,
                                const _Float16* __restrict__ Wp, int Cout,
                                const int* __restrict__ eidx,
                                float* __restrict__ Out, float outscale) {
  extern __shared__ _Float16 xs[];           // Cin*16 halves, fragment order
  int tile = blockIdx.y;
  int b = (tile * 16) / S_;
  int s0 = (tile * 16) % S_;
  int KT = Cin >> 5;
  for (int p = threadIdx.x; p < Cin * 16; p += blockDim.x) {
    int k = p >> 4, n = p & 15;
    int kt = k >> 5, kin = k & 31;
    xs[((kt * 32 + frag_lane(kin, n)) << 4) + frag_j(kin)] =
        (_Float16)X[((size_t)b * Cin + k) * S_ + s0 + n];
  }
  __syncthreads();
  int wave = threadIdx.x >> 5, lane = threadIdx.x & 31;
  int mt0 = blockIdx.x * 16 + wave * 2;      // m-tile index (units of 16 rows)
  int col = lane & 15, khalf = (lane >> 4) * 8;
  int myexp = eidx[tile * 16 + col];
  size_t fragstride = (size_t)KT * 512;      // halves per m-tile
  v8f c0s = {}, c1s = {};
  for (int e = 0; e < E_; ++e) {
    if (!__any(myexp == e)) continue;        // wave-uniform expert skip
    const _Float16* We = Wp + (size_t)e * (Cout >> 4) * fragstride;
    const _Float16* A0 = We + (size_t)(mt0 + 0) * fragstride + lane * 16;
    const _Float16* A1 = We + (size_t)(mt0 + 1) * fragstride + lane * 16;
    v8f c0 = {}, c1 = {};
    for (int kt = 0; kt < KT; ++kt) {
      if (kt + 1 < KT) __builtin_prefetch(A0 + (size_t)(kt + 1) * 512, 0, 0);
      v16h bv  = load_frag(&xs[(kt * 32 + lane) * 16]);
      v16h a0v = load_frag(A0 + (size_t)kt * 512);
      v16h a1v = load_frag(A1 + (size_t)kt * 512);
      c0 = __builtin_amdgcn_wmma_f32_16x16x32_f16(false, a0v, false, bv,
                                                  (short)0, c0, false, false);
      c1 = __builtin_amdgcn_wmma_f32_16x16x32_f16(false, a1v, false, bv,
                                                  (short)0, c1, false, false);
    }
    if (myexp == e) { c0s = c0; c1s = c1; }
  }
#pragma unroll
  for (int j = 0; j < 8; ++j) {
    int r0 = mt0 * 16 + j + khalf;
    int r1 = r0 + 16;
    Out[((size_t)b * Cout + r0) * S_ + s0 + col] = c0s[j] * outscale;
    Out[((size_t)b * Cout + r1) * S_ + s0 + col] = c1s[j] * outscale;
  }
}

// ---------------- cumsum over S per (b, channel), fused divisor/scale/shift -
__global__ void cumsum_kernel(const float* __restrict__ h0,
                              const float* __restrict__ divisor,
                              float* __restrict__ tmp1) {
  int row = blockIdx.x * (blockDim.x >> 5) + (threadIdx.x >> 5);
  int lane = threadIdx.x & 31;
  if (row >= B_ * I_) return;
  int b = row / I_, i = row % I_;
  const float* dep = h0 + ((size_t)b * 3 * I_ + i) * S_;
  const float* sc  = h0 + ((size_t)b * 3 * I_ + I_ + i) * S_;
  const float* sh  = h0 + ((size_t)b * 3 * I_ + 2 * I_ + i) * S_;
  float* o = tmp1 + ((size_t)b * I_ + i) * S_;
  float carry = 0.f;
  for (int s0 = 0; s0 < S_; s0 += 32) {
    float v = dep[s0 + lane];
#pragma unroll
    for (int d = 1; d < 32; d <<= 1) {
      float t = __shfl_up(v, d, 32);
      if (lane >= d) v += t;
    }
    float cum = carry + v;
    o[s0 + lane] = cum / divisor[s0 + lane] * sc[s0 + lane] + sh[s0 + lane];
    carry = __shfl(cum, 31, 32);
  }
}

// ---------------- channel norm (mean / scaled L2) + leaky relu --------------
__global__ void norm_kernel(const float* __restrict__ p0,
                            const float* __restrict__ p1,
                            const float* __restrict__ p2,
                            int cstride, int mode,
                            float* __restrict__ out) {
  __shared__ float red[256];
  int t = blockIdx.x; int b = t / S_, s = t % S_;
  float x[4]; float lsum = 0.f;
#pragma unroll
  for (int r = 0; r < 4; ++r) {
    int i = threadIdx.x + r * 256;
    size_t idx = ((size_t)b * cstride + i) * S_ + s;
    float v = p0[idx];
    if (mode) v = v * p1[idx] + p2[idx];
    x[r] = v; lsum += v;
  }
  red[threadIdx.x] = lsum; __syncthreads();
  for (int st = 128; st > 0; st >>= 1) {
    if (threadIdx.x < st) red[threadIdx.x] += red[threadIdx.x + st];
    __syncthreads();
  }
  float mean = red[0] * (1.0f / I_);
  __syncthreads();
  float lss = 0.f;
#pragma unroll
  for (int r = 0; r < 4; ++r) { float d = x[r] - mean; lss += d * d; }
  red[threadIdx.x] = lss; __syncthreads();
  for (int st = 128; st > 0; st >>= 1) {
    if (threadIdx.x < st) red[threadIdx.x] += red[threadIdx.x + st];
    __syncthreads();
  }
  float n = sqrtf(red[0]) * (1.0f / 32.0f);  // * I^-0.5
  float inv = 1.0f / (n + EPSN);
#pragma unroll
  for (int r = 0; r < 4; ++r) {
    int i = threadIdx.x + r * 256;
    float y = (x[r] - mean) * inv;
    y = y >= 0.f ? y : y * SLOPE_;
    out[((size_t)b * I_ + i) * S_ + s] = y;
  }
}

// ---------------- causal conv1d as WMMA GEMM (K = I*3 = 3072) ---------------
// G:[B,I,S] f32, Wp1 packed fragments (Cout=3I, Cin=3I kk=i*3+k), Out:[B,3I,S]
__global__ void conv_wmma_kernel(const float* __restrict__ G,
                                 const _Float16* __restrict__ Wp1,
                                 float* __restrict__ Out) {
  __shared__ _Float16 xs[128 * 16];          // one 128-K chunk, fragment order
  int tile = blockIdx.y;
  int b = (tile * 16) / S_, s0 = (tile * 16) % S_;
  int wave = threadIdx.x >> 5, lane = threadIdx.x & 31;
  int mt0 = blockIdx.x * 16 + wave * 2;
  int col = lane & 15, khalf = (lane >> 4) * 8;
  const int KTtot = (3 * I_) >> 5;           // 96
  size_t fragstride = (size_t)KTtot * 512;
  const _Float16* A0 = Wp1 + (size_t)(mt0 + 0) * fragstride + lane * 16;
  const _Float16* A1 = Wp1 + (size_t)(mt0 + 1) * fragstride + lane * 16;
  v8f c0 = {}, c1 = {};
  for (int cc = 0; cc < 3 * I_; cc += 128) {
    __syncthreads();
    for (int p = threadIdx.x; p < 128 * 16; p += blockDim.x) {
      int kk = cc + (p >> 4); int n = p & 15;
      int i = kk / 3, kc = kk - i * 3;
      int s = s0 + n - 2 + kc;               // causal pad: s<0 -> 0
      int klocal = p >> 4;
      int kt = klocal >> 5, kin = klocal & 31;
      xs[((kt * 32 + frag_lane(kin, n)) << 4) + frag_j(kin)] =
          (s >= 0) ? (_Float16)G[((size_t)b * I_ + i) * S_ + s] : (_Float16)0.0f;
    }
    __syncthreads();
    int ktbase = cc >> 5;
#pragma unroll
    for (int kt = 0; kt < 4; ++kt) {
      __builtin_prefetch(A0 + (size_t)(ktbase + kt + 1) * 512, 0, 0);
      v16h bv  = load_frag(&xs[(kt * 32 + lane) * 16]);
      v16h a0v = load_frag(A0 + (size_t)(ktbase + kt) * 512);
      v16h a1v = load_frag(A1 + (size_t)(ktbase + kt) * 512);
      c0 = __builtin_amdgcn_wmma_f32_16x16x32_f16(false, a0v, false, bv,
                                                  (short)0, c0, false, false);
      c1 = __builtin_amdgcn_wmma_f32_16x16x32_f16(false, a1v, false, bv,
                                                  (short)0, c1, false, false);
    }
  }
#pragma unroll
  for (int j = 0; j < 8; ++j) {
    int r0 = mt0 * 16 + j + khalf;
    int r1 = r0 + 16;
    Out[((size_t)b * (3 * I_) + r0) * S_ + s0 + col] = c0[j];
    Out[((size_t)b * (3 * I_) + r1) * S_ + s0 + col] = c1[j];
  }
}

// ---------------- loss scaffolding ------------------------------------------
__global__ void init_loss_kernel(float* lacc) {
  if (threadIdx.x < 16) lacc[threadIdx.x] = 0.f;
}
__global__ void finalize_loss_kernel(const float* __restrict__ lacc,
                                     float* __restrict__ out_loss) {
  if (threadIdx.x == 0) {
    const float invT = 1.0f / (float)TOK;
    float l = 0.f;
    for (int e = 0; e < E_; ++e) l += (lacc[e] * invT) * (lacc[4 + e] * invT);
    for (int e = 0; e < E_; ++e) l += (lacc[8 + e] * invT) * (lacc[12 + e] * invT);
    *out_loss = l;
  }
}

extern "C" void kernel_launch(void* const* d_in, const int* in_sizes, int n_in,
                              void* d_out, int out_size, void* d_ws, size_t ws_size,
                              hipStream_t stream) {
  const float* inp     = (const float*)d_in[0];
  const float* divisor = (const float*)d_in[1];
  const float* w0_gate = (const float*)d_in[2];
  const float* w0      = (const float*)d_in[3];
  const float* w1      = (const float*)d_in[4];
  const float* w2_gate = (const float*)d_in[5];
  const float* w2      = (const float*)d_in[6];
  float* out = (float*)d_out;

  char* ws = (char*)d_ws;
  size_t off = 0;
  float* lacc = (float*)(ws + off); off += 256;
  const size_t h0b = (size_t)B_ * 3 * I_ * S_ * sizeof(float);
  const size_t t1b = (size_t)B_ * I_ * S_ * sizeof(float);
  float* h0   = (float*)(ws + off); off += h0b;               // [B,3I,S]
  float* tmp1 = (float*)(ws + off); off += t1b;               // [B,I,S]
  float* g1   = (float*)(ws + off); off += t1b;               // [B,I,S]
  int*   idx0 = (int*)(ws + off); off += TOK * sizeof(int);
  int*   idx1 = (int*)(ws + off); off += TOK * sizeof(int);
  off = (off + 255) & ~(size_t)255;
  const size_t w0n = (size_t)E_ * F_ * 3 * I_;                // packed halves
  const size_t w1n = (size_t)3 * I_ * 3 * I_;
  const size_t w2n = (size_t)E_ * I_ * F_;
  _Float16* w0p = (_Float16*)(ws + off); off += w0n * 2;
  _Float16* w1p = (_Float16*)(ws + off); off += w1n * 2;
  _Float16* w2p = (_Float16*)(ws + off); off += w2n * 2;
  float* h2 = h0;    // conv output reuses h0 (h0 dead after cumsum)
  float* h3 = tmp1;  // norm2 output reuses tmp1 (dead after norm1)

  // ---- weight pre-pack (f16, fragment order) ----
  pack_weights_kernel<<<(int)(w0n / 256), 256, 0, stream>>>(w0, w0p, E_, F_, 3 * I_, 0);
  pack_weights_kernel<<<(int)(w1n / 256), 256, 0, stream>>>(w1, w1p, 1, 3 * I_, 3 * I_, 1);
  pack_weights_kernel<<<(int)(w2n / 256), 256, 0, stream>>>(w2, w2p, E_, I_, F_, 0);

  init_loss_kernel<<<1, 32, 0, stream>>>(lacc);

  // MoE0: gate + routed 512->3072 GEMM
  gate_kernel<<<TOK / 256, 256, E_ * F_ * sizeof(float), stream>>>(
      inp, F_, w0_gate, idx0, lacc);
  moe_wmma_kernel<<<dim3(3 * I_ / 256, TOK / 16), 256,
                    F_ * 16 * sizeof(_Float16), stream>>>(
      inp, F_, w0p, 3 * I_, idx0, h0, 1.0f);

  // cumsum/divisor/scale/shift, then channel norm + leaky
  cumsum_kernel<<<(B_ * I_) / 8, 256, 0, stream>>>(h0, divisor, tmp1);
  norm_kernel<<<TOK, 256, 0, stream>>>(tmp1, tmp1, tmp1, I_, 0, g1);

  // causal conv1d (3072x1024x3) as WMMA GEMM
  conv_wmma_kernel<<<dim3(3 * I_ / 256, TOK / 16), 256, 0, stream>>>(g1, w1p, h2);

  // s0*s1+sh, channel norm + leaky
  norm_kernel<<<TOK, 256, 0, stream>>>(
      h2, h2 + (size_t)I_ * S_, h2 + (size_t)2 * I_ * S_, 3 * I_, 1, h3);

  // MoE1: gate + routed 1024->512 GEMM straight into d_out
  gate_kernel<<<TOK / 256, 256, E_ * I_ * sizeof(float), stream>>>(
      h3, I_, w2_gate, idx1, lacc + 8);
  moe_wmma_kernel<<<dim3(F_ / 256, TOK / 16), 256,
                    I_ * 16 * sizeof(_Float16), stream>>>(
      h3, I_, w2p, F_, idx1, out, 1.0f /* INIT_SCALE */);

  finalize_loss_kernel<<<1, 32, 0, stream>>>(lacc, out + (size_t)B_ * F_ * S_);
}
